// GemmaAttention_14491219657103
// MI455X (gfx1250) — matrix operand order repaired and checked by reading, not compiled, not run on hardware
//
#include <hip/hip_runtime.h>
#include <stdint.h>
#include <stddef.h>

// ---------- CDNA5 (gfx1250) wave32 WMMA attention pipeline ----------
// B=1, S=2048, H=2048, NH=16, HD=128
#define SEQ    2048
#define HDIM   2048
#define NHEAD  16
#define HDHEAD 128

typedef __attribute__((ext_vector_type(16))) __bf16 v16bf;
typedef __attribute__((ext_vector_type(8)))  float  v8f;
typedef __attribute__((ext_vector_type(4)))  float  f32x4;
typedef __attribute__((ext_vector_type(4)))  unsigned int u32x4;

#define F_ROPE   1
#define F_TRANSC 2

// f32 -> bf16, round-half-up (bits + 0x8000): 2 VALU ops
__device__ __forceinline__ uint16_t f2bf(float f) {
  return (uint16_t)((__builtin_bit_cast(uint32_t, f) + 0x8000u) >> 16);
}

// two f32 -> packed bf16 pair: 2x v_add + 1x v_perm_b32
// v_perm_b32 D,S0,S1,sel: {S0,S1} byte pool, S1 = bytes 0-3.  sel 0x07060302
// picks [ua.b2, ua.b3, ub.b2, ub.b3] = hi16(ua) | hi16(ub)<<16.
__device__ __forceinline__ uint32_t pk2bf(float a, float b) {
  uint32_t ua = __builtin_bit_cast(uint32_t, a) + 0x8000u;
  uint32_t ub = __builtin_bit_cast(uint32_t, b) + 0x8000u;
  return __builtin_amdgcn_perm(ub, ua, 0x07060302u);
}

// ---- fragment loads from LDS (layouts per CDNA5 ISA 7.12.2) ----
// A 16x32 bf16: lanes 0-15 = rows; elems 0-7 K=kb..+7, elems 8-15 K=kb+16..+23, kb=(lane>>4)*8
// B 32x16 bf16 (stored transposed; row n contiguous in K): col=lane&15; elems 0-7 K=kb..+7, elems 8-15 K=kb+16..+23, kb=(lane>>4)*8
__device__ __forceinline__ v16bf frag_a(const uint16_t* base, int m0, int lane) {
  const uint16_t* p = base + (m0 + (lane & 15)) * 40 + (lane >> 4) * 8;
  union { u32x4 u[2]; v16bf v; } x;
  x.u[0] = *(const u32x4*)p;
  x.u[1] = *(const u32x4*)(p + 16);
  return x.v;
}
__device__ __forceinline__ v16bf frag_b(const uint16_t* base, int n0, int lane) {
  const uint16_t* p = base + (n0 + (lane & 15)) * 40 + (lane >> 4) * 8;
  union { u32x4 u[2]; v16bf v; } x;
  x.u[0] = *(const u32x4*)p;
  x.u[1] = *(const u32x4*)(p + 16);
  return x.v;
}

// =====================================================================
// Kernel 1: f32 operands, staged with perm-packed down-convert, ping-pong LDS.
//   C = alpha * A @ W^T ; FLAGS: F_ROPE (multiply rope[m][n&127]),
//   F_TRANSC (store C[n][m]); CBF16: bf16 output.
// 256 thr = 8 waves; block tile 128x128; BK=32; wave tile 64x32 (4x2 WMMA)
// =====================================================================
template<int FLAGS, bool CBF16>
__global__ __launch_bounds__(256)
void gemm_f32in(const float* __restrict__ A, int lda, long long aOffZ,
                const float* __restrict__ W, int ldw, long long wOffZ,
                void* __restrict__ Cv, int ldc, long long cOffZ,
                int K, float alpha, const float* __restrict__ rope)
{
  __shared__ uint16_t As[2][128][40];   // 80B row stride, 16B aligned
  __shared__ uint16_t Bs[2][128][40];

  const int t    = threadIdx.x;
  const int lane = t & 31;
  const int wave = t >> 5;
  const int m_wave = (wave & 1) * 64;
  const int n_wave = (wave >> 1) * 32;
  const int bm = blockIdx.y * 128;
  const int bn = blockIdx.x * 128;

  A += (long long)blockIdx.z * aOffZ;
  W += (long long)blockIdx.z * wOffZ;

  const int srow  = t >> 1;        // tile row staged by this thread
  const int shalf = (t & 1) * 16;  // 16-column half of BK=32

  const float* aRow = A + (size_t)(bm + srow) * lda + shalf;
  const float* bRow = W + (size_t)(bn + srow) * ldw + shalf;

  f32x4 av[4], bv[4];
  auto gload = [&](int step) {
    const f32x4* ap = (const f32x4*)(aRow + step * 32);
    const f32x4* bp = (const f32x4*)(bRow + step * 32);
#pragma unroll
    for (int i = 0; i < 4; ++i) { av[i] = ap[i]; bv[i] = bp[i]; }
  };
  auto lstore = [&](int buf) {
    u32x4 pa[2], pb[2];
#pragma unroll
    for (int i = 0; i < 4; ++i) {
      pa[i >> 1][(i & 1) * 2 + 0] = pk2bf(av[i].x, av[i].y);
      pa[i >> 1][(i & 1) * 2 + 1] = pk2bf(av[i].z, av[i].w);
      pb[i >> 1][(i & 1) * 2 + 0] = pk2bf(bv[i].x, bv[i].y);
      pb[i >> 1][(i & 1) * 2 + 1] = pk2bf(bv[i].z, bv[i].w);
    }
    *(u32x4*)&As[buf][srow][shalf    ] = pa[0];
    *(u32x4*)&As[buf][srow][shalf + 8] = pa[1];
    *(u32x4*)&Bs[buf][srow][shalf    ] = pb[0];
    *(u32x4*)&Bs[buf][srow][shalf + 8] = pb[1];
  };

  v8f acc[4][2] = {};
  auto compute = [&](int buf) {
    v16bf bfr[2];
#pragma unroll
    for (int j = 0; j < 2; ++j) bfr[j] = frag_b(&Bs[buf][0][0], n_wave + 16*j, lane);
#pragma unroll
    for (int i = 0; i < 4; ++i) {
      v16bf afr = frag_a(&As[buf][0][0], m_wave + 16*i, lane);
#pragma unroll
      for (int j = 0; j < 2; ++j)
        acc[i][j] = __builtin_amdgcn_wmma_f32_16x16x32_bf16(
            false, afr, false, bfr[j], (short)0, acc[i][j], false, false);
    }
  };

  const int nsteps = K >> 5;
  gload(0); lstore(0);
  __syncthreads();
  for (int s = 0; s < nsteps; ++s) {
    const int cur = s & 1;
    if (s + 1 < nsteps) gload(s + 1);      // global loads overlap compute
    compute(cur);
    if (s + 1 < nsteps) { lstore(cur ^ 1); __syncthreads(); }
  }

  // epilogue: VGPR r of a 16x16 C tile holds (M = r + 8*(lane>=16), N = lane&15)
  const int lhi  = (lane >> 4) * 8;
  const int lcol = lane & 15;
  float*    Cf = (float*)Cv    + (long long)blockIdx.z * cOffZ;
  uint16_t* Ch = (uint16_t*)Cv + (long long)blockIdx.z * cOffZ;
#pragma unroll
  for (int i = 0; i < 4; ++i) {
#pragma unroll
    for (int j = 0; j < 2; ++j) {
      const int tm = bm + m_wave + 16*i + lhi;
      const int tn = bn + n_wave + 16*j + lcol;
#pragma unroll
      for (int r = 0; r < 8; ++r) {
        const int m = tm + r;
        float v = acc[i][j][r] * alpha;
        if constexpr ((FLAGS & F_ROPE) != 0)
          v *= rope[(size_t)m * HDHEAD + (tn & (HDHEAD - 1))];
        const size_t idx = (FLAGS & F_TRANSC) ? ((size_t)tn * ldc + m)
                                              : ((size_t)m  * ldc + tn);
        if constexpr (CBF16) Ch[idx] = f2bf(v);
        else                 Cf[idx] = v;
      }
    }
  }
}

// =====================================================================
// Kernel 2: bf16 operands in global memory, ping-pong LDS staged with
// CDNA5 async DMA (global_load_async_to_lds_b128): the DMA for tile k+1
// runs concurrently with the WMMAs for tile k; one s_wait_asynccnt +
// barrier per step.  (IOFFSET adds to BOTH LDS and global addr, ISA 15.18.)
// =====================================================================
template<bool CBF16>
__global__ __launch_bounds__(256)
void gemm_bf16in(const uint16_t* __restrict__ A, int lda, long long aOffZ,
                 const uint16_t* __restrict__ W, int ldw, long long wOffZ,
                 void* __restrict__ Cv, int ldc, long long cOffZ,
                 int K, float alpha)
{
  __shared__ uint16_t As[2][128][40];
  __shared__ uint16_t Bs[2][128][40];

  const int t    = threadIdx.x;
  const int lane = t & 31;
  const int wave = t >> 5;
  const int m_wave = (wave & 1) * 64;
  const int n_wave = (wave >> 1) * 32;
  const int bm = blockIdx.y * 128;
  const int bn = blockIdx.x * 128;

  A += (long long)blockIdx.z * aOffZ;
  W += (long long)blockIdx.z * wOffZ;

  const int srow  = t >> 1;
  const int shalf = (t & 1) * 16;

  const uint16_t* aRow = A + (size_t)(bm + srow) * lda + shalf;
  const uint16_t* bRow = W + (size_t)(bn + srow) * ldw + shalf;

  // flat shared pointer truncated to 32 bits == LDS byte address (ISA 10.2)
  const uint32_t ldsA[2] = { (uint32_t)(size_t)&As[0][srow][shalf],
                             (uint32_t)(size_t)&As[1][srow][shalf] };
  const uint32_t ldsB[2] = { (uint32_t)(size_t)&Bs[0][srow][shalf],
                             (uint32_t)(size_t)&Bs[1][srow][shalf] };

  auto issue = [&](int step, int buf) {
    const uint16_t* ga = aRow + step * 32;
    const uint16_t* gb = bRow + step * 32;
    asm volatile("global_load_async_to_lds_b128 %0, %1, off"
                 :: "v"(ldsA[buf]), "v"(ga) : "memory");
    asm volatile("global_load_async_to_lds_b128 %0, %1, off offset:16"
                 :: "v"(ldsA[buf]), "v"(ga) : "memory");
    asm volatile("global_load_async_to_lds_b128 %0, %1, off"
                 :: "v"(ldsB[buf]), "v"(gb) : "memory");
    asm volatile("global_load_async_to_lds_b128 %0, %1, off offset:16"
                 :: "v"(ldsB[buf]), "v"(gb) : "memory");
  };

  v8f acc[4][2] = {};
  auto compute = [&](int buf) {
    v16bf bfr[2];
#pragma unroll
    for (int j = 0; j < 2; ++j) bfr[j] = frag_b(&Bs[buf][0][0], n_wave + 16*j, lane);
#pragma unroll
    for (int i = 0; i < 4; ++i) {
      v16bf afr = frag_a(&As[buf][0][0], m_wave + 16*i, lane);
#pragma unroll
      for (int j = 0; j < 2; ++j)
        acc[i][j] = __builtin_amdgcn_wmma_f32_16x16x32_bf16(
            false, afr, false, bfr[j], (short)0, acc[i][j], false, false);
    }
  };

  const int nsteps = K >> 5;
  issue(0, 0);
  asm volatile("s_wait_asynccnt 0x0" ::: "memory");
  __syncthreads();
  for (int s = 0; s < nsteps; ++s) {
    const int cur = s & 1;
    if (s + 1 < nsteps) issue(s + 1, cur ^ 1);   // DMA overlaps WMMA
    compute(cur);
    if (s + 1 < nsteps) {
      asm volatile("s_wait_asynccnt 0x0" ::: "memory");
      __syncthreads();
    }
  }

  const int lhi  = (lane >> 4) * 8;
  const int lcol = lane & 15;
  float*    Cf = (float*)Cv    + (long long)blockIdx.z * cOffZ;
  uint16_t* Ch = (uint16_t*)Cv + (long long)blockIdx.z * cOffZ;
#pragma unroll
  for (int i = 0; i < 4; ++i) {
#pragma unroll
    for (int j = 0; j < 2; ++j) {
      const int tm = bm + m_wave + 16*i + lhi;
      const int tn = bn + n_wave + 16*j + lcol;
#pragma unroll
      for (int r = 0; r < 8; ++r) {
        const size_t idx = (size_t)(tm + r) * ldc + tn;
        if constexpr (CBF16) Ch[idx] = f2bf(acc[i][j][r] * alpha);
        else                 Cf[idx] = acc[i][j][r] * alpha;
      }
    }
  }
}

// rope[m][d] = cos[m][d] + sin[m][d]  (reference's rotate_half is identity)
__global__ __launch_bounds__(256)
void rope_factor(const float* __restrict__ c, const float* __restrict__ s,
                 float* __restrict__ f)
{
  const size_t i = ((size_t)blockIdx.x * 256 + threadIdx.x) * 4;
  f32x4 a = *(const f32x4*)(c + i);
  f32x4 b = *(const f32x4*)(s + i);
  f32x4 r; r.x = a.x + b.x; r.y = a.y + b.y; r.z = a.z + b.z; r.w = a.w + b.w;
  *(f32x4*)(f + i) = r;
}

// elementwise f32 -> bf16 (8 elems/thread, perm-packed converts)
__global__ __launch_bounds__(256)
void cvt_f32_bf16(const float* __restrict__ src, uint16_t* __restrict__ dst)
{
  const size_t i = ((size_t)blockIdx.x * 256 + threadIdx.x) * 8;
  const f32x4* s = (const f32x4*)(src + i);
  f32x4 a = s[0], b = s[1];
  u32x4 p;
  p.x = pk2bf(a.x, a.y); p.y = pk2bf(a.z, a.w);
  p.z = pk2bf(b.x, b.y); p.w = pk2bf(b.z, b.w);
  *(u32x4*)(dst + i) = p;
}

// in-place row softmax (one 256-thread block per row)
__global__ __launch_bounds__(256)
void softmax_rows(float* __restrict__ P, int ncols)
{
  __shared__ float red[256];
  float* p = P + (size_t)blockIdx.x * ncols;
  const int t = threadIdx.x;

  float mx = -3.402823466e38f;
  for (int c = t; c < ncols; c += 256) mx = fmaxf(mx, p[c]);
  red[t] = mx; __syncthreads();
  for (int s = 128; s > 0; s >>= 1) { if (t < s) red[t] = fmaxf(red[t], red[t+s]); __syncthreads(); }
  mx = red[0]; __syncthreads();

  float sum = 0.f;
  for (int c = t; c < ncols; c += 256) { float e = __expf(p[c] - mx); p[c] = e; sum += e; }
  red[t] = sum; __syncthreads();
  for (int s = 128; s > 0; s >>= 1) { if (t < s) red[t] += red[t+s]; __syncthreads(); }
  const float inv = 1.0f / red[0];
  for (int c = t; c < ncols; c += 256) p[c] *= inv;
}

extern "C" void kernel_launch(void* const* d_in, const int* in_sizes, int n_in,
                              void* d_out, int out_size, void* d_ws, size_t ws_size,
                              hipStream_t stream)
{
  (void)in_sizes; (void)n_in; (void)out_size; (void)ws_size;
  const float* q_in = (const float*)d_in[0];   // (S,H)
  const float* k_in = (const float*)d_in[1];   // (S,H)
  const float* cosb = (const float*)d_in[2];   // (S,HD)
  const float* sinb = (const float*)d_in[3];   // (S,HD)
  const float* Wq   = (const float*)d_in[4];   // (H,H), K-major rows
  const float* Wk   = (const float*)d_in[5];
  const float* Wv   = (const float*)d_in[6];
  const float* Wo   = (const float*)d_in[7];

  float* outp  = (float*)d_out;                 // (S,H)    4M f32
  float* attnw = outp + (size_t)SEQ * HDIM;     // (NH,S,S) 64M f32

  // ws: Qbf 8M | Kbf 8M | ATbf 8M | Wobf 8M | VT(f32) 16M | rope(f32) 1M
  uint8_t* wsb = (uint8_t*)d_ws;
  uint16_t* Qbf  = (uint16_t*)(wsb);
  uint16_t* Kbf  = (uint16_t*)(wsb +  8u * 1024 * 1024);
  uint16_t* ATbf = (uint16_t*)(wsb + 16u * 1024 * 1024);
  uint16_t* Wobf = (uint16_t*)(wsb + 24u * 1024 * 1024);
  float*    VT   = (float*)   (wsb + 32u * 1024 * 1024);
  float*    RF   = (float*)   (wsb + 48u * 1024 * 1024);

  const dim3 blk(256);
  const dim3 g16(16, 16, 1);
  const float scale = 0.08838834764831845f;     // 128^-0.5

  // rope factor table (S*HD = 262144 elems, 4/thread)
  rope_factor<<<dim3(SEQ * HDHEAD / (256 * 4)), blk, 0, stream>>>(cosb, sinb, RF);
  // Wo -> bf16 for the async output projection
  cvt_f32_bf16<<<dim3((HDIM * HDIM) / (256 * 8)), blk, 0, stream>>>(Wo, Wobf);

  // q = query @ Wk^T (reference swaps Wq/Wk), degenerate RoPE -> bf16
  gemm_f32in<F_ROPE, true><<<g16, blk, 0, stream>>>(
      q_in, HDIM, 0, Wk, HDIM, 0, Qbf, HDIM, 0, HDIM, 1.0f, RF);
  // k = key @ Wq^T, degenerate RoPE -> bf16
  gemm_f32in<F_ROPE, true><<<g16, blk, 0, stream>>>(
      k_in, HDIM, 0, Wq, HDIM, 0, Kbf, HDIM, 0, HDIM, 1.0f, RF);
  // V^T = (key @ Wv^T)^T -> f32 (rows contiguous in K for the P@V stage)
  gemm_f32in<F_TRANSC, false><<<g16, blk, 0, stream>>>(
      k_in, HDIM, 0, Wv, HDIM, 0, VT, SEQ, 0, HDIM, 1.0f, nullptr);

  // scores[h] = (Q_h @ K_h^T) * scale -> f32 straight into d_out  [async staged]
  const dim3 gs(16, 16, NHEAD);
  gemm_bf16in<false><<<gs, blk, 0, stream>>>(
      Qbf, HDIM, HDHEAD, Kbf, HDIM, HDHEAD,
      attnw, SEQ, (long long)SEQ * SEQ, HDHEAD, scale);

  // softmax rows in place (NH*S rows of S)
  softmax_rows<<<dim3(NHEAD * SEQ), blk, 0, stream>>>(attnw, SEQ);

  // attn[h] = P_h @ V_h (A = f32 probs in d_out, W = VT f32) -> bf16 context
  const dim3 gpv(1, 16, NHEAD);
  gemm_f32in<0, true><<<gpv, blk, 0, stream>>>(
      attnw, SEQ, (long long)SEQ * SEQ, VT, SEQ, (long long)HDHEAD * SEQ,
      ATbf, HDIM, HDHEAD, SEQ, 1.0f, nullptr);

  // out = attn @ Wo^T (both bf16) [async staged] -> f32 d_out
  gemm_bf16in<false><<<g16, blk, 0, stream>>>(
      ATbf, HDIM, 0, Wobf, HDIM, 0, outp, HDIM, 0, HDIM, 1.0f);
}
